// MoEQLoRALinear_42073499631858
// MI455X (gfx1250) — matrix operand order repaired and checked by reading, not compiled, not run on hardware
//
#include <hip/hip_runtime.h>
#include <hip/hip_bf16.h>
#include <stdint.h>

typedef __attribute__((ext_vector_type(16))) __bf16 v16bf;
typedef __attribute__((ext_vector_type(8)))  float  v8f;

#define BATCH   4
#define SEQ     2048
#define MDIM    (BATCH*SEQ)   // 8192
#define INP     4096
#define OUTP    4096
#define NEXP    4
#define RANK    8
#define NER     (NEXP*RANK)   // 32
#define SCALE_F 2.0f          // alpha / r = 16/8

#define KTILES  (INP/32)      // 128 main K-tiles; tile 128 is the LoRA tile
#define AROW    40            // padded LDS row stride (elems) for 32-elem rows
#define BROW    40

// ---------- helpers ----------
__device__ __forceinline__ unsigned short f2bf(float f) {
    unsigned u = __float_as_uint(f);
    unsigned r = u + 0x7FFFu + ((u >> 16) & 1u);   // round-to-nearest-even
    return (unsigned short)(r >> 16);
}

union FragAB {
    uint4 u[2];
    v16bf v;
};

// async global->LDS copy of 16 bytes (ASYNCcnt-tracked, GV addressing)
__device__ __forceinline__ void async_cp16(unsigned lds_byte, const void* gaddr) {
    asm volatile("global_load_async_to_lds_b128 %0, %1, off"
                 :: "v"(lds_byte), "v"(gaddr)
                 : "memory");
}

// ---------- kernel 1: W fp32 -> bf16 ----------
__global__ void convert_w_kernel(const float* __restrict__ W,
                                 unsigned short* __restrict__ Wb) {
    int idx = (blockIdx.x * blockDim.x + threadIdx.x) * 4;   // 4 elems/thread
    if (idx >= OUTP * INP) return;
    float4 f = *reinterpret_cast<const float4*>(W + idx);
    uint2 p;
    p.x = (unsigned)f2bf(f.x) | ((unsigned)f2bf(f.y) << 16);
    p.y = (unsigned)f2bf(f.z) | ((unsigned)f2bf(f.w) << 16);
    *reinterpret_cast<uint2*>(Wb + idx) = p;
}

// ---------- kernel 2: BmT[o][e*8+r] = bf16(Bm[e][o][r]) ----------
__global__ void build_bmt_kernel(const float* __restrict__ Bm,
                                 unsigned short* __restrict__ BmT) {
    int idx = blockIdx.x * blockDim.x + threadIdx.x;
    if (idx >= OUTP * NER) return;
    int o  = idx / NER;
    int er = idx % NER;
    int e  = er >> 3, r = er & 7;
    float v = Bm[(size_t)e * OUTP * RANK + (size_t)o * RANK + r];
    BmT[idx] = f2bf(v);
}

// ---------- kernel 3: per-token router + proj + x->bf16 ----------
__global__ void router_proj_kernel(const float* __restrict__ x,
                                   const float* __restrict__ Wr,   // (E, IN)
                                   const float* __restrict__ A,    // (E*R, IN)
                                   unsigned short* __restrict__ Xb,
                                   unsigned short* __restrict__ Coef) {
    const int m    = blockIdx.x;
    const int tid  = threadIdx.x;
    const int lane = tid & 31;
    const int wave = tid >> 5;            // 0..7
    const float* xrow = x + (size_t)m * INP;

    float acc[4 + NER];
    #pragma unroll
    for (int t = 0; t < 4 + NER; ++t) acc[t] = 0.0f;

    #pragma unroll
    for (int j = 0; j < INP / 256; ++j) {
        int   i  = tid + j * 256;
        float xv = xrow[i];
        Xb[(size_t)m * INP + i] = f2bf(xv);
        #pragma unroll
        for (int e = 0; e < 4; ++e)
            acc[e] += xv * Wr[e * INP + i];
        #pragma unroll
        for (int k = 0; k < NER; ++k)
            acc[4 + k] += xv * A[(size_t)k * INP + i];
    }

    #pragma unroll
    for (int t = 0; t < 4 + NER; ++t) {
        #pragma unroll
        for (int off = 16; off > 0; off >>= 1)
            acc[t] += __shfl_xor(acc[t], off, 32);
    }

    __shared__ float sred[8][4 + NER];
    __shared__ float sfin[4 + NER];
    __shared__ float sgate[4];
    if (lane == 0) {
        #pragma unroll
        for (int t = 0; t < 4 + NER; ++t) sred[wave][t] = acc[t];
    }
    __syncthreads();
    if (tid < 4 + NER) {
        float s = 0.0f;
        #pragma unroll
        for (int w = 0; w < 8; ++w) s += sred[w][tid];
        sfin[tid] = s;
    }
    __syncthreads();
    if (tid == 0) {
        float mx = sfin[0];
        #pragma unroll
        for (int e = 1; e < 4; ++e) mx = fmaxf(mx, sfin[e]);
        float den = 0.0f, ex[4];
        #pragma unroll
        for (int e = 0; e < 4; ++e) { ex[e] = __expf(sfin[e] - mx); den += ex[e]; }
        float inv = 1.0f / den;
        #pragma unroll
        for (int e = 0; e < 4; ++e) sgate[e] = ex[e] * inv * SCALE_F;
    }
    __syncthreads();
    if (tid < NER) {
        int e = tid >> 3;
        Coef[(size_t)m * NER + tid] = f2bf(sgate[e] * sfin[4 + tid]);
    }
}

// ---------- kernel 4: bf16 WMMA GEMM, async-LDS double buffered ----------
// block: 256 thr = 8 waves (2 x 4). block tile 128(M) x 256(N); wave tile 64x64.
__global__ void __launch_bounds__(256)
gemm_kernel(const unsigned short* __restrict__ Xb,   // (M, IN)
            const unsigned short* __restrict__ Wb,   // (OUT, IN)
            const unsigned short* __restrict__ Coef, // (M, 32)
            const unsigned short* __restrict__ BmT,  // (OUT, 32)
            const float* __restrict__ bias,          // (OUT)
            float* __restrict__ out) {               // (M, OUT)
    const int tid  = threadIdx.x;
    const int lane = tid & 31;
    const int w    = tid >> 5;              // 0..7
    const int wm   = w >> 2;                // 0..1
    const int wn   = w & 3;                 // 0..3
    const int lr   = lane & 15;
    const int hi   = lane >> 4;             // 0 | 1

    const int mBlk = blockIdx.y * 128;
    const int nBlk = blockIdx.x * 256;

    // double-buffered staging tiles (rows padded to 40 elems = 80B stride)
    __shared__ __align__(16) unsigned short As[2][128 * AROW];  // 2 x 10 KB
    __shared__ __align__(16) unsigned short Bs[2][256 * BROW];  // 2 x 20 KB

    // ISA A-frag (16-bit 16x32): lanes 0-15: K 0..7 / 16..23; lanes 16-31: 8..15 / 24..31
    const unsigned kA = hi ? 8u : 0u;
    // ISA B-frag (32x16): lanes 0-15: K 0..7 / 16..23; lanes 16-31: 8..15 / 24..31
    const unsigned kB = hi ? 8u : 0u;

    // per-thread copy assignments: A tile = 512 x 16B, B tile = 1024 x 16B
    const int rA0 = tid >> 2,            sA0 = tid & 3;          // i = 0
    const int rA1 = (tid + 256) >> 2,    sA1 = tid & 3;          // i = 1
    // B rows: id = tid + i*256 -> row = id>>2, seg = id&3

    // issue one full tile fill (exactly 6 async ops per thread)
    auto fill = [&](int buf, int step) {
        const unsigned short* Ab;
        const unsigned short* Bb;
        unsigned aStr, bStr, kOff;
        if (step < KTILES) { Ab = Xb;   aStr = INP; Bb = Wb;  bStr = INP; kOff = (unsigned)step * 32u; }
        else               { Ab = Coef; aStr = NER; Bb = BmT; bStr = NER; kOff = 0u; }
        #pragma unroll
        for (int i = 0; i < 2; ++i) {
            int id  = tid + i * 256;
            int row = id >> 2, seg = id & 3;
            async_cp16((unsigned)(uintptr_t)&As[buf][row * AROW + seg * 8],
                       Ab + (size_t)(mBlk + row) * aStr + kOff + seg * 8);
        }
        #pragma unroll
        for (int i = 0; i < 4; ++i) {
            int id  = tid + i * 256;
            int row = id >> 2, seg = id & 3;
            async_cp16((unsigned)(uintptr_t)&Bs[buf][row * BROW + seg * 8],
                       Bb + (size_t)(nBlk + row) * bStr + kOff + seg * 8);
        }
    };

    v8f acc[4][4];
    #pragma unroll
    for (int mi = 0; mi < 4; ++mi)
        #pragma unroll
        for (int ni = 0; ni < 4; ++ni)
            acc[mi][ni] = (v8f){0.f, 0.f, 0.f, 0.f, 0.f, 0.f, 0.f, 0.f};

    fill(0, 0);

    for (int step = 0; step <= KTILES; ++step) {
        const int cur = step & 1;
        if (step < KTILES) {
            fill(cur ^ 1, step + 1);
            // 6 newer ops outstanding; async loads complete in order, so <=6
            // guarantees this wave's current-tile copies have landed
            asm volatile("s_wait_asynccnt 0x6" ::: "memory");
        } else {
            asm volatile("s_wait_asynccnt 0x0" ::: "memory");
        }
        __syncthreads();   // all waves' copies for 'cur' complete

        // A fragments resident (32 VGPRs), B fragments streamed
        FragAB a[4];
        #pragma unroll
        for (int mi = 0; mi < 4; ++mi) {
            const unsigned short* p = &As[cur][(wm * 64 + mi * 16 + lr) * AROW + kA];
            a[mi].u[0] = *reinterpret_cast<const uint4*>(p);
            a[mi].u[1] = *reinterpret_cast<const uint4*>(p + 16);
        }
        #pragma unroll
        for (int ni = 0; ni < 4; ++ni) {
            const unsigned short* p = &Bs[cur][(wn * 64 + ni * 16 + lr) * BROW + kB];
            FragAB b;
            b.u[0] = *reinterpret_cast<const uint4*>(p);
            b.u[1] = *reinterpret_cast<const uint4*>(p + 16);
            #pragma unroll
            for (int mi = 0; mi < 4; ++mi)
                acc[mi][ni] = __builtin_amdgcn_wmma_f32_16x16x32_bf16(
                    false, a[mi].v, false, b.v,
                    (short)0, acc[mi][ni], false, false);
        }
        __syncthreads();   // done reading 'cur' before it is refilled
    }

    // epilogue: add bias, store fp32
    #pragma unroll
    for (int ni = 0; ni < 4; ++ni) {
        const int col = nBlk + wn * 64 + ni * 16 + lr;
        const float bv = bias[col];
        #pragma unroll
        for (int mi = 0; mi < 4; ++mi) {
            #pragma unroll
            for (int j = 0; j < 8; ++j) {
                const int mrow = mBlk + wm * 64 + mi * 16 + j + hi * 8;
                out[(size_t)mrow * OUTP + col] = acc[mi][ni][j] + bv;
            }
        }
    }
}

// ---------- launch ----------
extern "C" void kernel_launch(void* const* d_in, const int* in_sizes, int n_in,
                              void* d_out, int out_size, void* d_ws, size_t ws_size,
                              hipStream_t stream) {
    const float* x  = (const float*)d_in[0];   // (B,S,IN)
    const float* W  = (const float*)d_in[1];   // (OUT,IN)
    const float* b  = (const float*)d_in[2];   // (OUT)
    const float* A  = (const float*)d_in[3];   // (E,R,IN)
    const float* Bm = (const float*)d_in[4];   // (E,OUT,R)
    const float* Wr = (const float*)d_in[5];   // (E,IN)
    float* out = (float*)d_out;

    char* ws = (char*)d_ws;
    unsigned short* Xb   = (unsigned short*)(ws);                          // 64 MiB
    unsigned short* Wb   = (unsigned short*)(ws + (size_t)MDIM * INP * 2); // 32 MiB
    unsigned short* Coef = (unsigned short*)(ws + (size_t)MDIM * INP * 2
                                                + (size_t)OUTP * INP * 2);
    unsigned short* BmT  = (unsigned short*)(ws + (size_t)MDIM * INP * 2
                                                + (size_t)OUTP * INP * 2
                                                + (size_t)MDIM * NER * 2);

    {
        int threads = (OUTP * INP) / 4;
        convert_w_kernel<<<(threads + 255) / 256, 256, 0, stream>>>(W, Wb);
    }
    {
        int threads = OUTP * NER;
        build_bmt_kernel<<<(threads + 255) / 256, 256, 0, stream>>>(Bm, BmT);
    }
    router_proj_kernel<<<MDIM, 256, 0, stream>>>(x, Wr, A, Xb, Coef);
    {
        dim3 grid(OUTP / 256, MDIM / 128);
        gemm_kernel<<<grid, 256, 0, stream>>>(Xb, Wb, Coef, BmT, b, out);
    }
}
